// TAGCN_14491219656876
// MI455X (gfx1250) — matrix operand order repaired and checked by reading, not compiled, not run on hardware
//
#include <hip/hip_runtime.h>
#include <hip/hip_bf16.h>

#define NN 100000
#define NE 3200000
#define ZSTRIDE 256   // padded concat stride for layer1 (4 blocks of 64)
#define HSTRIDE 128

typedef __attribute__((ext_vector_type(16))) _Float16 v16h;
typedef __attribute__((ext_vector_type(8)))  _Float16 v8h;
typedef __attribute__((ext_vector_type(8)))  float    v8f;

__device__ __forceinline__ void atomAddF32(float* p, float v) {
    // lowers to global_atomic_add_f32 on gfx1250 (relaxed, device scope)
    __hip_atomic_fetch_add(p, v, __ATOMIC_RELAXED, __HIP_MEMORY_SCOPE_AGENT);
}

// ---------------- utility kernels ----------------

__global__ __launch_bounds__(256) void zeroK(float* __restrict__ p, int n) {
    int i = blockIdx.x * 256 + threadIdx.x;
    if (i < n) p[i] = 0.0f;
}

// deg accumulated as float in dinv buffer
__global__ __launch_bounds__(256) void degK(const int* __restrict__ ei, float* __restrict__ deg) {
    int e = blockIdx.x * 256 + threadIdx.x;
    if (e < NE) atomAddF32(&deg[ei[NE + e]], 1.0f);
}

__global__ __launch_bounds__(256) void rsqrtK(float* __restrict__ dinv) {
    int i = blockIdx.x * 256 + threadIdx.x;
    if (i < NN) {
        float d = dinv[i];
        dinv[i] = (d > 0.0f) ? __frsqrt_rn(d) : 0.0f;
    }
}

// copy x[N,58] into z1[:, 0:58] (z1 pre-zeroed, stride 256)
__global__ __launch_bounds__(256) void scatterX(const float* __restrict__ x, float* __restrict__ z1) {
    int i = blockIdx.x * 256 + threadIdx.x;
    if (i < NN * 58) {
        int n = i / 58, c = i - n * 58;
        z1[(size_t)n * ZSTRIDE + c] = x[i];
    }
}

// ---------------- pre-pack W1 (232x128 f32) into B-fragment layout ----------------
// W1B[colTile(8)][ks(8)][lane(32)][half(16)] f16, zero-padded K: kp=ks*32+hi*16+i,
// blk=kp>>6, r=kp&63 valid iff r<58; column n = colTile*16 + (lane&15).
__global__ __launch_bounds__(256) void packW1(const float* __restrict__ W1, _Float16* __restrict__ W1B) {
    int t = blockIdx.x * 256 + threadIdx.x;      // 0 .. 8*8*32*16-1
    if (t >= 8 * 8 * 32 * 16) return;
    int i    = t & 15;
    int lane = (t >> 4) & 31;
    int ks   = (t >> 9) & 7;
    int ct   = t >> 12;
    int n      = ct * 16 + (lane & 15);
    int hiHalf = lane >> 4;
    int kp  = ks * 32 + hiHalf * 16 + i;
    int blk = kp >> 6, r = kp & 63;
    float v = (r < 58) ? W1[(size_t)(blk * 58 + r) * 128 + n] : 0.0f;
    W1B[t] = (_Float16)v;
}

// ---------------- SpMM (edge-parallel, wave per edge) ----------------

// layer1: 58 features inside z1; lane handles 2 columns
__global__ __launch_bounds__(256) void spmm58(const int* __restrict__ ei, const float* __restrict__ dinv,
                                              float* __restrict__ z1, int inOff, int outOff) {
    int wave = (blockIdx.x * 256 + threadIdx.x) >> 5;
    int lane = threadIdx.x & 31;
    if (wave >= NE) return;
    int s = ei[wave];
    int d = ei[NE + wave];
    float w = dinv[s] * dinv[d];
    const float* in  = z1 + (size_t)s * ZSTRIDE + inOff;
    float*       out = z1 + (size_t)d * ZSTRIDE + outOff;
    int c = lane * 2;
    if (c < 58) {
        float2 v = *(const float2*)(in + c);
        atomAddF32(out + c,     v.x * w);
        atomAddF32(out + c + 1, v.y * w);
    }
}

// layer2: 128 features; lane handles 4 columns
__global__ __launch_bounds__(256) void spmm128(const int* __restrict__ ei, const float* __restrict__ dinv,
                                               const float* __restrict__ gin, float* __restrict__ gout) {
    int wave = (blockIdx.x * 256 + threadIdx.x) >> 5;
    int lane = threadIdx.x & 31;
    if (wave >= NE) return;
    int s = ei[wave];
    int d = ei[NE + wave];
    float w = dinv[s] * dinv[d];
    const float* in  = gin  + (size_t)s * HSTRIDE;
    float*       out = gout + (size_t)d * HSTRIDE;
#pragma unroll
    for (int i = 0; i < 4; ++i) {
        int c = lane + 32 * i;
        atomAddF32(out + c, in[c] * w);
    }
}

// ---------------- WMMA GEMM: H = relu(z1[100000x256] @ W1pad[256x128] + b1) ----------------
// grid = 6250 blocks (16 rows each); 8 waves/block, one 16-col tile per wave.
// A tile (16x256) staged once through LDS as f16; B from pre-packed W1B.

__global__ __launch_bounds__(256) void gemm1(const float* __restrict__ z1, const _Float16* __restrict__ W1B,
                                             const float* __restrict__ b1, float* __restrict__ H) {
    __shared__ __attribute__((aligned(16))) _Float16 As[16 * 256];

    const int t      = threadIdx.x;
    const int wave   = t >> 5;             // 0..7 -> col tile
    const int lane   = t & 31;
    const int l15    = lane & 15;
    const int hiHalf = lane >> 4;          // 0 | 1
    const int row0   = blockIdx.x * 16;
    const int n      = wave * 16 + l15;    // output column for this lane

    // ---- cooperative stage: thread t handles (row = t>>4, 16-col chunk = t&15)
    {
        const int row   = t >> 4;
        const int chunk = (t & 15) * 16;
        const float4* src = (const float4*)(z1 + (size_t)(row0 + row) * ZSTRIDE + chunk);
        float4 f0 = src[0], f1 = src[1], f2 = src[2], f3 = src[3];
        v8h h0 = { (_Float16)f0.x, (_Float16)f0.y, (_Float16)f0.z, (_Float16)f0.w,
                   (_Float16)f1.x, (_Float16)f1.y, (_Float16)f1.z, (_Float16)f1.w };
        v8h h1 = { (_Float16)f2.x, (_Float16)f2.y, (_Float16)f2.z, (_Float16)f2.w,
                   (_Float16)f3.x, (_Float16)f3.y, (_Float16)f3.z, (_Float16)f3.w };
        *(v8h*)(As + row * 256 + chunk)     = h0;
        *(v8h*)(As + row * 256 + chunk + 8) = h1;
    }
    __syncthreads();

    v8f acc = {};
#pragma unroll
    for (int ks = 0; ks < 8; ++ks) {
        const int kBase = ks * 32;
        // A fragment: lane row = l15; halves 0..15 at K=kBase+hi*16+0..15
        const int k0 = kBase + hiHalf * 16;
        const int k1 = kBase + hiHalf * 16 + 8;
        v8h a0 = *(const v8h*)(As + l15 * 256 + k0);
        v8h a1 = *(const v8h*)(As + l15 * 256 + k1);
        v16h a = __builtin_shufflevector(a0, a1, 0,1,2,3,4,5,6,7,8,9,10,11,12,13,14,15);
        // B fragment: one contiguous 32B vector per lane from packed buffer
        v16h b = *(const v16h*)(W1B + (((size_t)(wave * 8 + ks) * 32 + lane) << 4));
        acc = __builtin_amdgcn_wmma_f32_16x16x32_f16(false, a, false, b, (short)0, acc, false, false);
    }
    // ---- store with bias + relu: VGPR r -> row = row0 + r + 8*hiHalf, col = n
    float bias = b1[n];
#pragma unroll
    for (int r = 0; r < 8; ++r) {
        int row = row0 + r + 8 * hiHalf;
        float v = acc[r] + bias;
        H[(size_t)row * HSTRIDE + n] = fmaxf(v, 0.0f);
    }
}

// ---------------- layer-2 matvec: out (+)= g[N,128] . W2blk[128] (+ b2) ----------------

__global__ __launch_bounds__(256) void dotW2(const float* __restrict__ g, const float* __restrict__ W2blk,
                                             const float* __restrict__ b2, float* __restrict__ out, int writeBias) {
    int wave = (blockIdx.x * 256 + threadIdx.x) >> 5;
    int lane = threadIdx.x & 31;
    if (wave >= NN) return;
    const float* row = g + (size_t)wave * HSTRIDE;
    float s = 0.0f;
#pragma unroll
    for (int i = 0; i < 4; ++i) {
        int c = lane + 32 * i;
        s += row[c] * W2blk[c];
    }
#pragma unroll
    for (int off = 16; off > 0; off >>= 1) s += __shfl_xor(s, off, 32);
    if (lane == 0) {
        if (writeBias) out[wave] = s + b2[0];
        else           out[wave] += s;
    }
}

// ---------------- launch ----------------

extern "C" void kernel_launch(void* const* d_in, const int* in_sizes, int n_in,
                              void* d_out, int out_size, void* d_ws, size_t ws_size,
                              hipStream_t stream) {
    const float* x  = (const float*)d_in[0];
    const int*   ei = (const int*)d_in[1];     // [2, NE] src row then dst row
    const float* W1 = (const float*)d_in[2];   // [232,128]
    const float* b1 = (const float*)d_in[3];   // [128]
    const float* W2 = (const float*)d_in[4];   // [512,1]
    const float* b2 = (const float*)d_in[5];   // [1]
    float* out = (float*)d_out;

    // workspace layout (floats). z1 region is reused as G1|G2 after gemm1.
    float*    wsf  = (float*)d_ws;
    float*    dinv = wsf;                                   // NN
    float*    z1   = wsf + NN;                              // NN*256
    float*    H    = z1 + (size_t)NN * ZSTRIDE;             // NN*128
    _Float16* W1B  = (_Float16*)(H + (size_t)NN * HSTRIDE); // 8*8*32*16 halves (64 KB)
    float*    G1   = z1;                                    // NN*128 (aliases dead z1)
    float*    G2   = z1 + (size_t)NN * HSTRIDE;             // NN*128

    const int T = 256;
    const int edgeThreadBlocks = (NE + T - 1) / T;           // thread-per-edge
    const int edgeWaveBlocks   = (NE * 32) / T;              // wave-per-edge = 400000
    const int nodeWaveBlocks   = (NN * 32 + T - 1) / T;      // wave-per-node = 12500
    const int nBlocksN         = (NN + T - 1) / T;
    const int zElems           = NN * ZSTRIDE;
    const int hElems           = NN * HSTRIDE;

    // 0) pack W1 into WMMA B-fragment layout (reused by all 6250 GEMM blocks)
    packW1<<<(8 * 8 * 32 * 16) / T, T, 0, stream>>>(W1, W1B);

    // 1) normalization weights
    zeroK<<<nBlocksN, T, 0, stream>>>(dinv, NN);
    degK<<<edgeThreadBlocks, T, 0, stream>>>(ei, dinv);
    rsqrtK<<<nBlocksN, T, 0, stream>>>(dinv);

    // 2) layer-1 propagation into padded concat buffer z1
    zeroK<<<(zElems + T - 1) / T, T, 0, stream>>>(z1, zElems);
    scatterX<<<(NN * 58 + T - 1) / T, T, 0, stream>>>(x, z1);
    spmm58<<<edgeWaveBlocks, T, 0, stream>>>(ei, dinv, z1, 0,   64);
    spmm58<<<edgeWaveBlocks, T, 0, stream>>>(ei, dinv, z1, 64,  128);
    spmm58<<<edgeWaveBlocks, T, 0, stream>>>(ei, dinv, z1, 128, 192);

    // 3) WMMA GEMM + bias + relu -> H
    gemm1<<<NN / 16, T, 0, stream>>>(z1, W1B, b1, H);

    // 4) layer-2: hop-0 dot, then 3 SpMM(128) + dot hops (z1 space reused)
    dotW2<<<nodeWaveBlocks, T, 0, stream>>>(H, W2 + 0,   b2, out, 1);

    zeroK<<<(hElems + T - 1) / T, T, 0, stream>>>(G1, hElems);
    spmm128<<<edgeWaveBlocks, T, 0, stream>>>(ei, dinv, H, G1);
    dotW2<<<nodeWaveBlocks, T, 0, stream>>>(G1, W2 + 128, b2, out, 0);

    zeroK<<<(hElems + T - 1) / T, T, 0, stream>>>(G2, hElems);
    spmm128<<<edgeWaveBlocks, T, 0, stream>>>(ei, dinv, G1, G2);
    dotW2<<<nodeWaveBlocks, T, 0, stream>>>(G2, W2 + 256, b2, out, 0);

    zeroK<<<(hElems + T - 1) / T, T, 0, stream>>>(G1, hElems);
    spmm128<<<edgeWaveBlocks, T, 0, stream>>>(ei, dinv, G2, G1);
    dotW2<<<nodeWaveBlocks, T, 0, stream>>>(G1, W2 + 384, b2, out, 0);
}
